// GroupConvMLP_50311246905403
// MI455X (gfx1250) — matrix operand order repaired and checked by reading, not compiled, not run on hardware
//
#include <hip/hip_runtime.h>
#include <math.h>

// ---------------------------------------------------------------------------
// Fused grouped MLP for MI455X (gfx1250, wave32, WMMA f16 16x16x32, f32 accum)
//   x[B,128] -> per-group 16->512->512->512->16 with swish, fp32 out.
// Weights pre-swizzled to WMMA B-fragment layout in f16, kb-major so each
// wave's per-k panel (8 fragments) is one contiguous 8KB block (imm-offset
// loads, 2 prefetches). 128-row batch tile per WG, single LDS h buffer
// updated in place (outputs staged in registers across a barrier).
// ---------------------------------------------------------------------------

typedef _Float16 h16;
typedef __attribute__((ext_vector_type(16))) _Float16 v16h;
typedef __attribute__((ext_vector_type(8)))  float    v8f;
typedef __attribute__((ext_vector_type(2)))  _Float16 v2h;
typedef __attribute__((ext_vector_type(2)))  __fp16   v2fp;
typedef __attribute__((ext_vector_type(4)))  unsigned int u32x4;
typedef __attribute__((ext_vector_type(4)))  float    f32x4;

#define G_    8
#define S_    16
#define H_    512
#define MT    128          // batch rows per workgroup
#define RS    520          // LDS row stride in halves (8-half pad -> conflict-free b128)
#define NT_H  (H_ / 16)    // 32 n-tiles
#define KT_H  (H_ / 32)    // 16 k-steps of 32

// workspace layout (in halves): swizzled f16 weights, kb-major fragments:
//   frag(g, kb, nb) at ((g*KT + kb)*NT + nb)*512
#define OFF_W1 0
#define LEN_W1 (G_ * NT_H * 1 * 512)       // 131072
#define OFF_W2 (OFF_W1 + LEN_W1)
#define LEN_W23 (G_ * NT_H * KT_H * 512)   // 2097152
#define OFF_W3 (OFF_W2 + LEN_W23)
#define OFF_W4 (OFF_W3 + LEN_W23)
#define LEN_W4 (G_ * 1 * KT_H * 512)       // 65536

#define KB_STRIDE ((size_t)NT_H * 512)     // halves between kb panels (32KB)

// --------------------------- device helpers -------------------------------

__device__ __forceinline__ float sigmoid_fast(float z) {
#if __has_builtin(__builtin_amdgcn_tanhf)
  return 0.5f * (1.0f + __builtin_amdgcn_tanhf(0.5f * z));     // v_tanh_f32 (gfx1250)
#elif __has_builtin(__builtin_amdgcn_tanh_f32)
  return 0.5f * (1.0f + __builtin_amdgcn_tanh_f32(0.5f * z));
#else
  return __builtin_amdgcn_rcpf(1.0f + __builtin_amdgcn_exp2f(-1.44269504089f * z));
#endif
}

__device__ __forceinline__ v2h pack2(float a, float b) {
#if __has_builtin(__builtin_amdgcn_cvt_pkrtz)
  union { v2fp p; v2h h; } u;
  u.p = __builtin_amdgcn_cvt_pkrtz(a, b);                      // v_cvt_pk_rtz_f16_f32
  return u.h;
#else
  v2h r; r.x = (_Float16)a; r.y = (_Float16)b; return r;
#endif
}

__device__ __forceinline__ v8f wmma_f16(v16h a, v16h b, v8f c) {
  // 8 args: (neg_a, A, neg_b, B, c_mod, C, reuse_a, reuse_b)
  return __builtin_amdgcn_wmma_f32_16x16x32_f16(false, a, false, b, (short)0, c,
                                                false, false);
}

// B-fragment: 1KB contiguous; lane L holds col n=L%16, K=(L/16)*8+e (e<8), 16+(L/16)*8+e-8 (e>=8) -> 32B/lane
__device__ __forceinline__ v16h load_bfrag(const h16* __restrict__ p) {
  union { u32x4 q[2]; v16h v; } u;
  u.q[0] = *(const u32x4*)(p);
  u.q[1] = *(const u32x4*)(p + 8);
  return u.v;
}

// A-fragment from LDS (row-major h): lane L row=L%16; chunk0 = K[(hi?8:0)..+8),
// chunk1 = K[16+(hi?8:0)..+8) (= +32 bytes).
__device__ __forceinline__ v16h load_afrag_lds(const h16* p) {
  union { u32x4 q[2]; v16h v; } u;
  u.q[0] = *(const u32x4*)(p);
  u.q[1] = *(const u32x4*)(p + 16);
  return u.v;
}

// bias + swish + pack f16, then barrier / in-place store / barrier.
__device__ __forceinline__ void epilogue_swish_store(
    v8f (&acc)[4][8], const float* __restrict__ bias, float sp, float inv_div,
    h16* __restrict__ hb, int g, int msub0, int nb0, int n_lo, int rsel)
{
  v2h st[4][8][4];
  const float* bp = bias + g * H_ + nb0 * 16 + n_lo;
#pragma unroll
  for (int i = 0; i < 4; i++) {
#pragma unroll
    for (int nt = 0; nt < 8; nt++) {
      float bb = bp[nt * 16];
#pragma unroll
      for (int j = 0; j < 4; j++) {
        float v0 = acc[i][nt][2 * j]     + bb;
        float v1 = acc[i][nt][2 * j + 1] + bb;
        float y0 = v0 * sigmoid_fast(v0 * sp) * inv_div;
        float y1 = v1 * sigmoid_fast(v1 * sp) * inv_div;
        st[i][nt][j] = pack2(y0, y1);
      }
    }
  }
  __syncthreads();   // all waves done reading h for this layer
  h16* hbase = hb + (msub0 * 16 + rsel * 8) * RS + nb0 * 16 + n_lo;
#pragma unroll
  for (int i = 0; i < 4; i++) {
#pragma unroll
    for (int nt = 0; nt < 8; nt++) {
#pragma unroll
      for (int j = 0; j < 4; j++) {
        hbase[(i * 16 + 2 * j    ) * RS + nt * 16] = st[i][nt][j][0];
        hbase[(i * 16 + 2 * j + 1) * RS + nt * 16] = st[i][nt][j][1];
      }
    }
  }
  __syncthreads();
}

// dense 512x512 layer: acc[4][8] += h(128x512, LDS) * W^T (swizzled f16, L2)
__device__ __forceinline__ void dense_layer(
    v8f (&acc)[4][8], const h16* __restrict__ hb, const h16* __restrict__ Wf,
    int g, int msub0, int nb0, int lane)
{
  const int n_lo = lane & 15;
  const int rsel = lane >> 4;
  const v8f vz = {0.f, 0.f, 0.f, 0.f, 0.f, 0.f, 0.f, 0.f};
#pragma unroll
  for (int i = 0; i < 4; i++)
#pragma unroll
    for (int nt = 0; nt < 8; nt++) acc[i][nt] = vz;

  // per-wave bases: B panel (contiguous 8KB per kb) and 4 LDS A rows
  const h16* wbase = Wf + (((size_t)g * KT_H) * NT_H + nb0) * 512 + (size_t)lane * 16;
  const h16* arow[4];
#pragma unroll
  for (int i = 0; i < 4; i++)
    arow[i] = hb + ((msub0 + i) * 16 + n_lo) * RS + rsel * 8;

  for (int kb = 0; kb < KT_H; kb++) {
    const h16* bp = wbase + (size_t)kb * KB_STRIDE;
    // prefetch next 8KB panel with two lane-spread lines
    {
      const h16* pn = wbase + (size_t)((kb + 1 < KT_H) ? kb + 1 : kb) * KB_STRIDE
                            - (size_t)lane * 16;           // panel origin
      __builtin_prefetch(pn + (size_t)lane * 64, 0, 1);    // 32 x 128B = 4KB
      __builtin_prefetch(pn + 2048 + (size_t)lane * 64, 0, 1);
    }
    v16h a[4];
#pragma unroll
    for (int i = 0; i < 4; i++)
      a[i] = load_afrag_lds(arow[i] + kb * 32);
#pragma unroll
    for (int nt = 0; nt < 8; nt++) {
      v16h bfr = load_bfrag(bp + nt * 512);
#pragma unroll
      for (int i = 0; i < 4; i++) acc[i][nt] = wmma_f16(a[i], bfr, acc[i][nt]);
    }
  }
}

// --------------------------- main fused kernel ----------------------------

__global__ __launch_bounds__(256, 1)
void fused_groupmlp(const float* __restrict__ x,
                    const float* __restrict__ b1, const float* __restrict__ beta1,
                    const float* __restrict__ b2, const float* __restrict__ beta2,
                    const float* __restrict__ b3, const float* __restrict__ beta3,
                    const float* __restrict__ b4,
                    const h16* __restrict__ ws,
                    float* __restrict__ out)
{
  __shared__ h16 hbuf[MT * RS];   // 133,120 bytes

  const int g    = blockIdx.y;
  const int row0 = blockIdx.x * MT;
  const int tid  = (int)threadIdx.x;
  const int lane = tid & 31;
  const int wave = tid >> 5;
  const int n_lo = lane & 15;
  const int rsel = lane >> 4;
  const int msub0 = (wave & 1) * 4;   // 2 M-halves  x 4 subtiles of 16 rows
  const int nb0   = (wave >> 1) * 8;  // 4 N-quarters x 8 tiles of 16 cols

  const h16* w1s = ws + OFF_W1;
  const h16* w2s = ws + OFF_W2;
  const h16* w3s = ws + OFF_W3;
  const h16* w4s = ws + OFF_W4;

  const float inv_div = (1.0f / 1.1f);
  const float sp1 = logf(1.0f + expf(beta1[g]));
  const float sp2 = logf(1.0f + expf(beta2[g]));
  const float sp3 = logf(1.0f + expf(beta3[g]));

  v8f acc[4][8];
  const v8f vz = {0.f, 0.f, 0.f, 0.f, 0.f, 0.f, 0.f, 0.f};

  // ---- Layer 1: h = swish(x * W1^T + b1), K=16 zero-padded to 32 ----
  {
    v16h a[4];
#pragma unroll
    for (int i = 0; i < 4; i++) {
      const float* xp = x + (size_t)(row0 + (msub0 + i) * 16 + n_lo) * (G_ * S_)
                          + g * S_ + rsel * 8;
      f32x4 f0 = *(const f32x4*)(xp);
      f32x4 f1 = *(const f32x4*)(xp + 4);
      v16h av;
      av[0] = (h16)f0.x; av[1] = (h16)f0.y; av[2] = (h16)f0.z; av[3] = (h16)f0.w;
      av[4] = (h16)f1.x; av[5] = (h16)f1.y; av[6] = (h16)f1.z; av[7] = (h16)f1.w;
#pragma unroll
      for (int t = 8; t < 16; t++) av[t] = (h16)0.0f;   // K pad
      a[i] = av;
    }
#pragma unroll
    for (int i = 0; i < 4; i++)
#pragma unroll
      for (int nt = 0; nt < 8; nt++) acc[i][nt] = vz;
    const h16* wb = w1s + ((size_t)(g * NT_H + nb0)) * 512 + (size_t)lane * 16;
#pragma unroll
    for (int nt = 0; nt < 8; nt++) {
      v16h bfr = load_bfrag(wb + nt * 512);
#pragma unroll
      for (int i = 0; i < 4; i++) acc[i][nt] = wmma_f16(a[i], bfr, acc[i][nt]);
    }
    epilogue_swish_store(acc, b1, sp1, inv_div, hbuf, g, msub0, nb0, n_lo, rsel);
  }

  // ---- Layer 2 & 3: h = swish(h * W^T + b), in-place via staged epilogue ----
  dense_layer(acc, hbuf, w2s, g, msub0, nb0, lane);
  epilogue_swish_store(acc, b2, sp2, inv_div, hbuf, g, msub0, nb0, n_lo, rsel);

  dense_layer(acc, hbuf, w3s, g, msub0, nb0, lane);
  epilogue_swish_store(acc, b3, sp3, inv_div, hbuf, g, msub0, nb0, n_lo, rsel);

  // ---- Layer 4: out = h * W4^T + b4  (N=16, K=512), one M-subtile per wave ----
  {
    v8f c4 = vz;
    const h16* ab = hbuf + (wave * 16 + n_lo) * RS + rsel * 8;
    const h16* wb = w4s + ((size_t)(g * KT_H)) * 512 + (size_t)lane * 16;
    for (int kb = 0; kb < KT_H; kb++) {
      v16h a   = load_afrag_lds(ab + kb * 32);
      v16h bfr = load_bfrag(wb + (size_t)kb * 512);
      c4 = wmma_f16(a, bfr, c4);
    }
    float bb = b4[g * S_ + n_lo];
#pragma unroll
    for (int j = 0; j < 8; j++) {
      int row = row0 + wave * 16 + rsel * 8 + j;
      out[(size_t)row * (G_ * S_) + g * S_ + n_lo] = c4[j] + bb;  // FACTOR == 1.0
    }
  }
}

// ------------------- weight swizzle to WMMA-B f16 layout -------------------
// out[((g*KT + kb)*NT + nb)*512 + lane*16 + e] = W[g, nb*16 + n, kb*32 + (e<8 ? (lane/16)*8 + e : 16 + (lane/16)*8 + e-8)]
// (zero-padded where k >= InC; W shaped [G, OutC, InC] row-major)
__global__ void swizzle_w(const float* __restrict__ W, h16* __restrict__ Wo,
                          int OutC, int InC, int KT)
{
  int idx = blockIdx.x * blockDim.x + (int)threadIdx.x;
  int NT = OutC >> 4;
  int total = G_ * NT * KT * 512;
  if (idx >= total) return;
  int e    = idx & 15;
  int lane = (idx >> 4) & 31;
  int nb   = (idx >> 9) % NT;
  int rem  = idx / (512 * NT);
  int kb   = rem % KT;
  int g    = rem / KT;
  int n    = lane & 15;
  int k    = kb * 32 + ((e < 8) ? ((lane >> 4) << 3) + e : 16 + ((lane >> 4) << 3) + (e - 8));
  float v  = 0.0f;
  if (k < InC) v = W[((size_t)g * OutC + nb * 16 + n) * (size_t)InC + k];
  Wo[idx] = (h16)v;
}

// --------------------------------- launch ----------------------------------

extern "C" void kernel_launch(void* const* d_in, const int* in_sizes, int n_in,
                              void* d_out, int out_size, void* d_ws, size_t ws_size,
                              hipStream_t stream) {
  const float* x     = (const float*)d_in[0];
  const float* W1    = (const float*)d_in[1];
  const float* b1    = (const float*)d_in[2];
  const float* beta1 = (const float*)d_in[3];
  const float* W2    = (const float*)d_in[4];
  const float* b2    = (const float*)d_in[5];
  const float* beta2 = (const float*)d_in[6];
  const float* W3    = (const float*)d_in[7];
  const float* b3    = (const float*)d_in[8];
  const float* beta3 = (const float*)d_in[9];
  const float* W4    = (const float*)d_in[10];
  const float* b4    = (const float*)d_in[11];
  float* out = (float*)d_out;
  h16*   ws  = (h16*)d_ws;

  const int Bsz = in_sizes[0] / (G_ * S_);

  // re-swizzle every call (deterministic; d_ws contents fully rewritten)
  {
    int t1 = G_ * NT_H * 1 * 512;
    swizzle_w<<<(t1 + 255) / 256, 256, 0, stream>>>(W1, ws + OFF_W1, H_, S_, 1);
    int t2 = G_ * NT_H * KT_H * 512;
    swizzle_w<<<(t2 + 255) / 256, 256, 0, stream>>>(W2, ws + OFF_W2, H_, H_, KT_H);
    swizzle_w<<<(t2 + 255) / 256, 256, 0, stream>>>(W3, ws + OFF_W3, H_, H_, KT_H);
    int t4 = G_ * 1 * KT_H * 512;
    swizzle_w<<<(t4 + 255) / 256, 256, 0, stream>>>(W4, ws + OFF_W4, S_, H_, KT_H);
  }

  dim3 grid(Bsz / MT, G_);
  fused_groupmlp<<<grid, dim3(256), 0, stream>>>(
      x, b1, beta1, b2, beta2, b3, beta3, b4, ws, out);
}
